// TripletLoss_11227044511928
// MI455X (gfx1250) — compile-verified, matrix operand order checked by reading, not run on hardware
//
#include <hip/hip_runtime.h>

#define BS 4096
#define DIM 512
#define MARGIN 0.3f

#define TILE 128     // block tile (M and N)
#define KC 32        // K-chunk per LDS stage
#define LSTR 40      // LDS row stride in bf16 elems (80B -> conflict-free b128 reads)
#define NSTAGES (DIM / KC)

typedef __bf16 bf16_t;
typedef bf16_t v16bf __attribute__((ext_vector_type(16)));
typedef float  v8f   __attribute__((ext_vector_type(8)));

static __device__ __forceinline__ v8f wmma_bf16(v16bf a, v16bf b, v8f c) {
    // D = A(16x32) * B(32x16) + C, fp32 accumulate
    return __builtin_amdgcn_wmma_f32_16x16x32_bf16(
        /*neg_a=*/false, a, /*neg_b=*/false, b,
        /*c_mod=*/(short)0, c, /*reuse_a=*/false, /*reuse_b=*/false);
}

// CDNA5 async copy: global -> LDS, 16 bytes per lane, tracked by ASYNCcnt.
// LDS address = low 32 bits of the generic shared pointer (flat-LDS truncation).
static __device__ __forceinline__ void async_b128(bf16_t* lds_ptr, const bf16_t* gptr) {
    unsigned lds_off = (unsigned)(unsigned long long)lds_ptr;
    asm volatile("global_load_async_to_lds_b128 %0, %1, off"
                 :: "v"(lds_off), "v"(gptr)
                 : "memory");
}

static __device__ __forceinline__ void wait_async_le8() {
    asm volatile("s_wait_asynccnt 0x8" ::: "memory");
}
static __device__ __forceinline__ void wait_async_0() {
    asm volatile("s_wait_asynccnt 0x0" ::: "memory");
}

// Load a 16x32 bf16 fragment from row-major LDS tile.
// Lane layout (ISA 7.12.2, 16-bit A 16x32): row = lane&15,
// K = (lane>>4)*8 + [0..7] for elems 0..7, and +16 for elems 8..15.
static __device__ __forceinline__ v16bf load_frag(const bf16_t* rowbase, int g) {
    union { uint4 u[2]; v16bf v; } f;
    f.u[0] = *(const uint4*)(rowbase + g * 8);
    f.u[1] = *(const uint4*)(rowbase + g * 8 + 16);
    return f.v;
}

// ---------------- Kernel 1: norms + bf16 hi/lo split -----------------------
__global__ __launch_bounds__(256) void prep_kernel(
    const float* __restrict__ sketch, const float* __restrict__ photo,
    bf16_t* __restrict__ Phi, bf16_t* __restrict__ Plo,
    bf16_t* __restrict__ Shi, bf16_t* __restrict__ Slo,
    float* __restrict__ p_sq, float* __restrict__ s_sq, float* __restrict__ pos)
{
    __shared__ float red[3][256];
    const int r = blockIdx.x;
    const int t = threadIdx.x;

    float s2 = 0.f, p2 = 0.f, d2 = 0.f;
    #pragma unroll
    for (int k = t; k < DIM; k += 256) {
        const int idx = r * DIM + k;
        float s = sketch[idx];
        float p = photo[idx];
        s2 += s * s;
        p2 += p * p;
        float d = s - p;
        d2 += d * d;
        bf16_t sh = (bf16_t)s;
        bf16_t ph = (bf16_t)p;
        Shi[idx] = sh;
        Slo[idx] = (bf16_t)(s - (float)sh);
        Phi[idx] = ph;
        Plo[idx] = (bf16_t)(p - (float)ph);
    }
    red[0][t] = s2; red[1][t] = p2; red[2][t] = d2;
    __syncthreads();
    #pragma unroll
    for (int off = 128; off > 0; off >>= 1) {
        if (t < off) {
            red[0][t] += red[0][t + off];
            red[1][t] += red[1][t + off];
            red[2][t] += red[2][t + off];
        }
        __syncthreads();
    }
    if (t == 0) {
        s_sq[r] = red[0][0];
        p_sq[r] = red[1][0];
        pos[r]  = sqrtf(red[2][0]);
    }
}

// ---------------- Kernel 2: WMMA GEMM + fused triplet epilogue -------------
__global__ __launch_bounds__(256) void triplet_gemm_kernel(
    const bf16_t* __restrict__ Phi, const bf16_t* __restrict__ Plo,
    const bf16_t* __restrict__ Shi, const bf16_t* __restrict__ Slo,
    const float* __restrict__ p_sq, const float* __restrict__ s_sq,
    const float* __restrict__ pos, float* __restrict__ partials)
{
    __shared__ bf16_t sAhi[2][TILE * LSTR];
    __shared__ bf16_t sAlo[2][TILE * LSTR];
    __shared__ bf16_t sBhi[2][TILE * LSTR];
    __shared__ bf16_t sBlo[2][TILE * LSTR];
    __shared__ float  sP[TILE], sS[TILE], sPos[TILE];
    __shared__ float  sRed[256];

    const int iBase = blockIdx.y * TILE;   // photo rows  (M)
    const int jBase = blockIdx.x * TILE;   // sketch rows (N)
    const int t     = threadIdx.x;
    const int wave  = t >> 5;
    const int lane  = t & 31;
    const int waveM = wave >> 1;           // 0..3 -> M offset waveM*32
    const int waveN = wave & 1;            // 0..1 -> N offset waveN*64
    const int g     = lane >> 4;           // lane half
    const int lr    = lane & 15;

    // Per-thread async-copy assignment: 2 rows x 4 arrays = 8 b128 per stage.
    const int row0 = t >> 2;               // 0..63
    const int kc   = (t & 3) << 3;         // 0,8,16,24 (bf16 elems)

    if (t < TILE) {
        sP[t]   = p_sq[iBase + t];
        sS[t]   = s_sq[jBase + t];
        sPos[t] = pos[jBase + t];
    }

    v8f acc[2][4];
    #pragma unroll
    for (int m = 0; m < 2; ++m)
        #pragma unroll
        for (int n = 0; n < 4; ++n)
            acc[m][n] = (v8f){0.f, 0.f, 0.f, 0.f, 0.f, 0.f, 0.f, 0.f};

    // ---- async double-buffered pipeline over K -----------------------------
    #define ISSUE_STAGE(buf, kk)                                                \
        do {                                                                    \
            _Pragma("unroll")                                                   \
            for (int h = 0; h < 2; ++h) {                                       \
                const int row = row0 + h * 64;                                  \
                const int ga  = (iBase + row) * DIM + (kk) + kc;                \
                const int gb  = (jBase + row) * DIM + (kk) + kc;                \
                const int li  = row * LSTR + kc;                                \
                async_b128(&sAhi[buf][li], Phi + ga);                           \
                async_b128(&sAlo[buf][li], Plo + ga);                           \
                async_b128(&sBhi[buf][li], Shi + gb);                           \
                async_b128(&sBlo[buf][li], Slo + gb);                           \
            }                                                                   \
        } while (0)

    ISSUE_STAGE(0, 0);   // prologue: stage 0 -> buf 0

    for (int s = 0; s < NSTAGES; ++s) {
        const int buf = s & 1;
        if (s + 1 < NSTAGES) {
            ISSUE_STAGE(buf ^ 1, (s + 1) * KC);   // prefetch next chunk
            wait_async_le8();                     // oldest 8 (current buf) done
        } else {
            wait_async_0();
        }
        __syncthreads();                          // all waves' async writes visible

        v16bf ah[2], al[2], bh[4], bl[4];
        #pragma unroll
        for (int m = 0; m < 2; ++m) {
            const int row = waveM * 32 + m * 16 + lr;
            ah[m] = load_frag(&sAhi[buf][row * LSTR], g);
            al[m] = load_frag(&sAlo[buf][row * LSTR], g);
        }
        #pragma unroll
        for (int n = 0; n < 4; ++n) {
            const int col = waveN * 64 + n * 16 + lr;
            bh[n] = load_frag(&sBhi[buf][col * LSTR], g);
            bl[n] = load_frag(&sBlo[buf][col * LSTR], g);
        }
        #pragma unroll
        for (int m = 0; m < 2; ++m)
            #pragma unroll
            for (int n = 0; n < 4; ++n) {
                acc[m][n] = wmma_bf16(ah[m], bh[n], acc[m][n]);  // hi*hi
                acc[m][n] = wmma_bf16(ah[m], bl[n], acc[m][n]);  // hi*lo
                acc[m][n] = wmma_bf16(al[m], bh[n], acc[m][n]);  // lo*hi
            }
        __syncthreads();   // WAR: everyone done reading buf before re-fill
    }

    // Epilogue: neg_sq -> dist -> hinge, diagonal masked, per-thread sum.
    // C/D layout: VGPR r, lanes 0-15: M=r, N=lane; lanes 16-31: M=r+8, N=lane-16.
    float lsum = 0.f;
    #pragma unroll
    for (int m = 0; m < 2; ++m)
        #pragma unroll
        for (int n = 0; n < 4; ++n)
            #pragma unroll
            for (int r = 0; r < 8; ++r) {
                const int il = waveM * 32 + m * 16 + r + (g ? 8 : 0);
                const int jl = waveN * 64 + n * 16 + lr;
                const float cross = acc[m][n][r];
                const float nsq = sP[il] + sS[jl] - 2.f * cross;
                const float nd  = sqrtf(fmaxf(nsq, 0.f));
                float tr = sPos[jl] - nd + MARGIN;
                if (iBase + il == jBase + jl) tr = 0.f;    // diagonal mask
                lsum += fmaxf(tr, 0.f);
            }

    sRed[t] = lsum;
    __syncthreads();
    #pragma unroll
    for (int off = 128; off > 0; off >>= 1) {
        if (t < off) sRed[t] += sRed[t + off];
        __syncthreads();
    }
    if (t == 0) partials[blockIdx.y * gridDim.x + blockIdx.x] = sRed[0];
}

// ---------------- Kernel 3: deterministic final reduction ------------------
__global__ __launch_bounds__(256) void reduce_kernel(
    const float* __restrict__ partials, float* __restrict__ out, int n)
{
    __shared__ float red[256];
    const int t = threadIdx.x;
    float s = 0.f;
    for (int i = t; i < n; i += 256) s += partials[i];
    red[t] = s;
    __syncthreads();
    #pragma unroll
    for (int off = 128; off > 0; off >>= 1) {
        if (t < off) red[t] += red[t + off];
        __syncthreads();
    }
    if (t == 0) out[0] = red[0];
}

extern "C" void kernel_launch(void* const* d_in, const int* in_sizes, int n_in,
                              void* d_out, int out_size, void* d_ws, size_t ws_size,
                              hipStream_t stream) {
    const float* sketch = (const float*)d_in[0];
    const float* photo  = (const float*)d_in[1];

    const size_t MAT = (size_t)BS * DIM;           // 2,097,152 bf16 elems
    char* ws = (char*)d_ws;
    bf16_t* Phi = (bf16_t*)ws;
    bf16_t* Plo = Phi + MAT;
    bf16_t* Shi = Plo + MAT;
    bf16_t* Slo = Shi + MAT;
    float* freg     = (float*)(ws + 4 * MAT * sizeof(bf16_t));
    float* p_sq     = freg;
    float* s_sq     = freg + BS;
    float* pos      = freg + 2 * BS;
    float* partials = freg + 3 * BS;

    prep_kernel<<<BS, 256, 0, stream>>>(sketch, photo, Phi, Plo, Shi, Slo,
                                        p_sq, s_sq, pos);

    dim3 grid(BS / TILE, BS / TILE);               // 32 x 32
    triplet_gemm_kernel<<<grid, 256, 0, stream>>>(Phi, Plo, Shi, Slo,
                                                  p_sq, s_sq, pos, partials);

    reduce_kernel<<<1, 256, 0, stream>>>(partials, (float*)d_out,
                                         (BS / TILE) * (BS / TILE));
}
